// GQAttention_37254546325673
// MI455X (gfx1250) — matrix operand order repaired and checked by reading, not compiled, not run on hardware
//
#include <hip/hip_runtime.h>
#include <hip/hip_bf16.h>
#include <stdint.h>

// ---- problem constants ----
#define Bc   4
#define Tc   2048
#define DIMc 2048
#define Hc   16
#define KVHc 4
#define HDc  128
#define KVDc 512

typedef __attribute__((ext_vector_type(16))) __bf16 v16bf;
typedef __attribute__((ext_vector_type(8)))  float  v8f;

#define DEV static __device__ __forceinline__

DEV __bf16 f2bf(float f) {
  union { float f; uint32_t u; } in; in.f = f;
  uint32_t r = in.u + 0x7FFFu + ((in.u >> 16) & 1u);  // round-to-nearest-even
  union { uint16_t s; __bf16 b; } out; out.s = (uint16_t)(r >> 16);
  return out.b;
}

// A fragment (16x32) as this file loads it, contiguous k, same order as its B loads:
// lane<16: row=lane, K in {kb+0..15}; lane>=16: row=lane-16, K in {kb+16..31}
// rowptr points at (this lane's row, kb).
DEV v16bf load_a(const __bf16* rowptr, int lane) {
  const int klo = (lane & 16) ? 16 : 0;
  union { v16bf v; uint4 u[2]; } r;
  r.u[0] = *(const uint4*)(rowptr + klo);
  r.u[1] = *(const uint4*)(rowptr + klo + 8);
  return r.v;
}

DEV v8f wmma_bf16(v16bf a, v16bf b, v8f c) {
  return __builtin_amdgcn_wmma_f32_16x16x32_bf16(false, a, false, b, (short)0, c, false, false);
}

// ---- kernel 0a: W[K][N] f32 -> WT[N][K] bf16 ----
__global__ void k_transpose_cvt(const float* __restrict__ W, __bf16* __restrict__ WT,
                                int K, int N) {
  long long idx = (long long)blockIdx.x * blockDim.x + threadIdx.x;
  long long total = (long long)K * N;
  if (idx >= total) return;
  int k = (int)(idx / N), n = (int)(idx % N);
  WT[(size_t)n * K + k] = f2bf(W[idx]);
}

// ---- kernel 0b: x f32 -> bf16, 4 elements/thread ----
__global__ void k_cvt_x(const float* __restrict__ x, __bf16* __restrict__ xb, long long total4) {
  long long i = (long long)blockIdx.x * blockDim.x + threadIdx.x;
  if (i >= total4) return;
  float4 f = ((const float4*)x)[i];
  union { __bf16 b[4]; uint2 u; } o;
  o.b[0] = f2bf(f.x); o.b[1] = f2bf(f.y); o.b[2] = f2bf(f.z); o.b[3] = f2bf(f.w);
  ((uint2*)xb)[i] = o.u;
}

// ---- kernel 1: fused QKV projection + bias + RoPE ----
// grid: x = 24 (16 Q heads, 4 K heads, 4 V heads), y = 256 row tiles of 32
// block: 256 threads = 8 waves; wave w owns cols [w*16, w*16+16) x 32 rows (2 row groups)
__global__ __launch_bounds__(256) void k_qkv(
    const __bf16* __restrict__ xb, const float* __restrict__ cosp, const float* __restrict__ sinp,
    const __bf16* __restrict__ wqT, const __bf16* __restrict__ wkT, const __bf16* __restrict__ wvT,
    const float* __restrict__ bq, const float* __restrict__ bk, const float* __restrict__ bv,
    __bf16* __restrict__ q_ws, __bf16* __restrict__ k_ws, __bf16* __restrict__ vt_ws)
{
  __shared__ __attribute__((aligned(16))) __bf16 xs[2][32][64];  // 2 x 4KB double buffer
  __shared__ __attribute__((aligned(16))) float  pre[32][128];   // 16KB epilogue staging

  const int tid  = threadIdx.x;
  const int lane = tid & 31, wid = tid >> 5;
  const int bx   = blockIdx.x;
  const int row0 = blockIdx.y * 32;
  const int bidx = row0 / Tc;
  const int t0   = row0 - bidx * Tc;

  const __bf16* WT; const float* bias; int colbase;
  if (bx < 16)      { WT = wqT; bias = bq; colbase = bx * 128; }
  else if (bx < 20) { WT = wkT; bias = bk; colbase = (bx - 16) * 128; }
  else              { WT = wvT; bias = bv; colbase = (bx - 20) * 128; }

  const int n      = lane & 15;
  const int klo_b  = (lane & 16) ? 16 : 0;
  const int hi8    = (lane & 16) ? 8 : 0;
  const int wavecol = wid * 16;
  const __bf16* bcol = WT + (size_t)(colbase + wavecol + n) * DIMc;

  // async stage: 32 rows x 64 cols bf16 = 4KB; 256 threads x 16B each
  const int sr = tid >> 3, sc = (tid & 7) * 8;
  auto stage = [&](int bi, int kb) {
    unsigned long long g =
        (unsigned long long)(uintptr_t)(xb + (size_t)(row0 + sr) * DIMc + kb + sc);
    unsigned l = (unsigned)(uintptr_t)&xs[bi][sr][sc];
    asm volatile("global_load_async_to_lds_b128 %0, %1, off" :: "v"(l), "v"(g) : "memory");
  };

  v8f acc[2] = {};
  stage(0, 0);
  asm volatile("s_wait_asynccnt 0x0" ::: "memory");
  __syncthreads();

  for (int kb = 0; kb < DIMc; kb += 64) {
    const int bi = (kb >> 6) & 1;
    if (kb + 64 < DIMc) stage(bi ^ 1, kb + 64);
    #pragma unroll
    for (int kc = 0; kc < 2; ++kc) {
      v16bf b = *(const v16bf*)(bcol + kb + kc * 32 + klo_b);
      #pragma unroll
      for (int rg = 0; rg < 2; ++rg) {
        v16bf a = load_a(&xs[bi][rg * 16 + (lane & 15)][kc * 32], lane);
        acc[rg] = wmma_bf16(a, b, acc[rg]);
      }
    }
    asm volatile("s_wait_asynccnt 0x0" ::: "memory");
    __syncthreads();
  }

  // bias + stage accumulators to LDS for the RoPE pairing
  {
    const float biasv = bias[colbase + wavecol + n];
    #pragma unroll
    for (int rg = 0; rg < 2; ++rg) {
      #pragma unroll
      for (int r = 0; r < 8; ++r) {
        pre[rg * 16 + r + hi8][wavecol + n] = acc[rg][r] + biasv;
      }
    }
  }
  __syncthreads();

  const bool isV = (bx >= 20);
  for (int e = tid; e < 32 * 128; e += 256) {
    int r = e >> 7, d = e & 127;
    int t = t0 + r;
    if (isV) {
      int kvh = bx - 20;
      vt_ws[((size_t)((bidx * KVHc + kvh) * HDc + d)) * Tc + t] = f2bf(pre[r][d]);
    } else {
      float val;
      if (d < 64) {
        float c = cosp[(size_t)t * 64 + d], s = sinp[(size_t)t * 64 + d];
        val = pre[r][2 * d] * c - pre[r][2 * d + 1] * s;
      } else {
        int dd = d - 64;
        float c = cosp[(size_t)t * 64 + dd], s = sinp[(size_t)t * 64 + dd];
        val = pre[r][2 * dd] * s + pre[r][2 * dd + 1] * c;
      }
      if (bx < 16) {
        q_ws[((size_t)((bidx * Hc + bx) * Tc + t)) * HDc + d] = f2bf(val);
      } else {
        int kvh = bx - 16;
        k_ws[((size_t)((bidx * KVHc + kvh) * Tc + t)) * HDc + d] = f2bf(val);
      }
    }
  }
}

// ---- kernel 2: flash attention, one wave per (b, h, 16-query tile) ----
__global__ __launch_bounds__(32) void k_attn(
    const __bf16* __restrict__ q_ws, const __bf16* __restrict__ k_ws,
    const __bf16* __restrict__ vt_ws, __bf16* __restrict__ ao)
{
  __shared__ __attribute__((aligned(16))) __bf16 pst[16][32];

  const int lane = threadIdx.x & 31;
  const int q0   = blockIdx.x * 16;
  const int bh   = blockIdx.y;
  const int bidx = bh >> 4, h = bh & 15, kvh = h >> 2;

  const __bf16* qbase = q_ws + ((size_t)(bidx * Hc + h) * Tc) * HDc;
  const __bf16* kbase = k_ws + ((size_t)(bidx * KVHc + kvh) * Tc) * HDc;
  const __bf16* vtb   = vt_ws + ((size_t)(bidx * KVHc + kvh) * HDc) * Tc;

  const int n     = lane & 15;
  const int klo_b = (lane & 16) ? 16 : 0;
  const int hi8   = (lane & 16) ? 8 : 0;

  v16bf aq[4];
  const __bf16* qrow = qbase + (size_t)(q0 + (lane & 15)) * HDc;
  #pragma unroll
  for (int dc = 0; dc < 4; ++dc) aq[dc] = load_a(qrow + dc * 32, lane);

  v8f o[8] = {};
  float rowmax[8], rowsum[8];
  #pragma unroll
  for (int r = 0; r < 8; ++r) { rowmax[r] = -1e30f; rowsum[r] = 0.0f; }

  const float scale = 0.08838834764831845f; // 1/sqrt(128)

  for (int kb = 0; kb < Tc; kb += 32) {
    if (kb + 32 < Tc) {
      __builtin_prefetch(kbase + (size_t)(kb + 32 + n) * HDc, 0, 1);
      __builtin_prefetch(vtb + (size_t)n * Tc + kb + 32, 0, 1);
    }
    v8f s0 = {}, s1 = {};
    #pragma unroll
    for (int dc = 0; dc < 4; ++dc) {
      v16bf b0 = *(const v16bf*)(kbase + (size_t)(kb + n)      * HDc + dc * 32 + klo_b);
      v16bf b1 = *(const v16bf*)(kbase + (size_t)(kb + 16 + n) * HDc + dc * 32 + klo_b);
      s0 = wmma_bf16(aq[dc], b0, s0);
      s1 = wmma_bf16(aq[dc], b1, s1);
    }

    float mx[8];
    #pragma unroll
    for (int r = 0; r < 8; ++r) {
      s0[r] *= scale; s1[r] *= scale;
      mx[r] = fmaxf(s0[r], s1[r]);
    }
    #pragma unroll
    for (int off = 1; off < 16; off <<= 1) {
      #pragma unroll
      for (int r = 0; r < 8; ++r) mx[r] = fmaxf(mx[r], __shfl_xor(mx[r], off, 32));
    }
    float al[8], p0[8], p1[8], rs[8];
    #pragma unroll
    for (int r = 0; r < 8; ++r) {
      float mnew = fmaxf(rowmax[r], mx[r]);
      al[r] = __expf(rowmax[r] - mnew);
      rowmax[r] = mnew;
      p0[r] = __expf(s0[r] - mnew);
      p1[r] = __expf(s1[r] - mnew);
      rs[r] = p0[r] + p1[r];
    }
    #pragma unroll
    for (int off = 1; off < 16; off <<= 1) {
      #pragma unroll
      for (int r = 0; r < 8; ++r) rs[r] += __shfl_xor(rs[r], off, 32);
    }
    #pragma unroll
    for (int r = 0; r < 8; ++r) rowsum[r] = rowsum[r] * al[r] + rs[r];
    #pragma unroll
    for (int c = 0; c < 8; ++c) {
      #pragma unroll
      for (int r = 0; r < 8; ++r) o[c][r] *= al[r];
    }

    // C-layout -> A-layout for P via LDS (in-wave DS ordering)
    #pragma unroll
    for (int r = 0; r < 8; ++r) {
      int m = r + hi8;
      pst[m][n]      = f2bf(p0[r]);
      pst[m][16 + n] = f2bf(p1[r]);
    }
    v16bf ap = load_a(&pst[lane & 15][0], lane);

    #pragma unroll
    for (int c = 0; c < 8; ++c) {
      v16bf bvf = *(const v16bf*)(vtb + (size_t)(c * 16 + n) * Tc + kb + klo_b);
      o[c] = wmma_bf16(ap, bvf, o[c]);
    }
  }

  float inv[8];
  #pragma unroll
  for (int r = 0; r < 8; ++r) inv[r] = 1.0f / rowsum[r];
  #pragma unroll
  for (int c = 0; c < 8; ++c) {
    #pragma unroll
    for (int r = 0; r < 8; ++r) {
      int m = r + hi8;
      size_t row = (size_t)bidx * Tc + q0 + m;
      ao[row * DIMc + (size_t)h * HDc + c * 16 + n] = f2bf(o[c][r] * inv[r]);
    }
  }
}

// ---- kernel 3: out = attn @ Wo + bo (f32 output) ----
// grid: x = 16 col blocks of 128, y = 256 row tiles of 32; block = 8 waves
__global__ __launch_bounds__(256) void k_oproj(
    const __bf16* __restrict__ ao, const __bf16* __restrict__ woT,
    const float* __restrict__ bo, float* __restrict__ out)
{
  const int tid  = threadIdx.x;
  const int lane = tid & 31, wid = tid >> 5;
  const int row0 = blockIdx.y * 32;
  const int col0 = blockIdx.x * 128 + wid * 16;
  const int n     = lane & 15;
  const int klo_b = (lane & 16) ? 16 : 0;
  const int hi8   = (lane & 16) ? 8 : 0;

  const __bf16* arow0 = ao  + (size_t)(row0 + (lane & 15)) * DIMc;
  const __bf16* arow1 = ao  + (size_t)(row0 + 16 + (lane & 15)) * DIMc;
  const __bf16* bcol  = woT + (size_t)(col0 + n) * DIMc;

  v8f acc0 = {}, acc1 = {};
  for (int kb = 0; kb < DIMc; kb += 32) {
    v16bf b = *(const v16bf*)(bcol + kb + klo_b);
    acc0 = wmma_bf16(load_a(arow0 + kb, lane), b, acc0);
    acc1 = wmma_bf16(load_a(arow1 + kb, lane), b, acc1);
  }
  const float biasv = bo[col0 + n];
  #pragma unroll
  for (int r = 0; r < 8; ++r) {
    int m = r + hi8;
    out[(size_t)(row0 + m) * DIMc + col0 + n]      = acc0[r] + biasv;
    out[(size_t)(row0 + 16 + m) * DIMc + col0 + n] = acc1[r] + biasv;
  }
}

extern "C" void kernel_launch(void* const* d_in, const int* in_sizes, int n_in,
                              void* d_out, int out_size, void* d_ws, size_t ws_size,
                              hipStream_t stream) {
  const float* x    = (const float*)d_in[0];
  const float* cosp = (const float*)d_in[1];
  const float* sinp = (const float*)d_in[2];
  const float* Wq   = (const float*)d_in[3];
  const float* bq   = (const float*)d_in[4];
  const float* Wk   = (const float*)d_in[5];
  const float* bk   = (const float*)d_in[6];
  const float* Wv   = (const float*)d_in[7];
  const float* bv   = (const float*)d_in[8];
  const float* Wo   = (const float*)d_in[9];
  const float* bo   = (const float*)d_in[10];
  float* out = (float*)d_out;
  (void)in_sizes; (void)n_in; (void)out_size; (void)ws_size;

  char* ws = (char*)d_ws;
  size_t off = 0;
  auto wsalloc = [&](size_t bytes) -> char* {
    char* p = ws + off; off += (bytes + 255) & ~(size_t)255; return p;
  };
  __bf16* wqT  = (__bf16*)wsalloc((size_t)DIMc * DIMc * 2);
  __bf16* wkT  = (__bf16*)wsalloc((size_t)KVDc * DIMc * 2);
  __bf16* wvT  = (__bf16*)wsalloc((size_t)KVDc * DIMc * 2);
  __bf16* woT  = (__bf16*)wsalloc((size_t)DIMc * DIMc * 2);
  __bf16* q_ws = (__bf16*)wsalloc((size_t)Bc * Hc  * Tc * HDc * 2);
  __bf16* k_ws = (__bf16*)wsalloc((size_t)Bc * KVHc * Tc * HDc * 2);
  __bf16* vt_ws= (__bf16*)wsalloc((size_t)Bc * KVHc * HDc * Tc * 2);
  __bf16* ao   = (__bf16*)wsalloc((size_t)Bc * Tc * DIMc * 2);
  __bf16* xb   = (__bf16*)wsalloc((size_t)Bc * Tc * DIMc * 2);

  {
    const int bs = 256;
    long long tot = (long long)DIMc * DIMc;
    k_transpose_cvt<<<(unsigned)((tot + bs - 1) / bs), bs, 0, stream>>>(Wq, wqT, DIMc, DIMc);
    tot = (long long)DIMc * KVDc;
    k_transpose_cvt<<<(unsigned)((tot + bs - 1) / bs), bs, 0, stream>>>(Wk, wkT, DIMc, KVDc);
    k_transpose_cvt<<<(unsigned)((tot + bs - 1) / bs), bs, 0, stream>>>(Wv, wvT, DIMc, KVDc);
    tot = (long long)DIMc * DIMc;
    k_transpose_cvt<<<(unsigned)((tot + bs - 1) / bs), bs, 0, stream>>>(Wo, woT, DIMc, DIMc);

    long long tot4 = (long long)Bc * Tc * DIMc / 4;
    k_cvt_x<<<(unsigned)((tot4 + bs - 1) / bs), bs, 0, stream>>>(x, xb, tot4);
  }

  k_qkv<<<dim3(24, (Bc * Tc) / 32), 256, 0, stream>>>(
      xb, cosp, sinp, wqT, wkT, wvT, bq, bk, bv, q_ws, k_ws, vt_ws);

  k_attn<<<dim3(Tc / 16, Bc * Hc), 32, 0, stream>>>(q_ws, k_ws, vt_ws, ao);

  k_oproj<<<dim3(DIMc / 128, (Bc * Tc) / 32), 256, 0, stream>>>(ao, woT, bo, out);
}
